// MultiModalHGAT_18640158065205
// MI455X (gfx1250) — matrix operand order repaired and checked by reading, compiled, not run on hardware
//
#include <hip/hip_runtime.h>
#include <math.h>

typedef __attribute__((ext_vector_type(16))) _Float16 v16h;
typedef __attribute__((ext_vector_type(8)))  float    v8f;

__device__ __forceinline__ float leakyf(float x, float s) { return x >= 0.f ? x : s * x; }

// pack 4 f32 -> 4 f16 and store as one 8-byte LDS write
__device__ __forceinline__ void st4h(_Float16* dst, float4 v)
{
  union { _Float16 h[4]; unsigned long long u; } t;
  t.h[0] = (_Float16)v.x; t.h[1] = (_Float16)v.y;
  t.h[2] = (_Float16)v.z; t.h[3] = (_Float16)v.w;
  *reinterpret_cast<unsigned long long*>(dst) = t.u;
}

// =====================================================================
// WMMA GEMM: C[M,N] = act(A[M,K] @ B[K,N] + bias)   (f32 in/out, f16 MACs)
// AMODE 0: A row-major.  AMODE 1: implicit im2col (3x3x3 conv) gather from
//          im_src [batch][cin][in][in][in]; rows -> (b,oz,oy,ox), cols -> (ci,kz,ky,kx).
// BT 0: B row-major [K,N].  BT 1: B stored [N,K].
// Block: 256 threads = 8 waves; block tile 64x64, BK=32, double-buffered LDS,
// register-staged (global_load_b128 fast path for interior tiles).
// Wave (wm=wid>>1, wn=wid&1) computes a 16x32 strip = 2 WMMAs / k-step.
// =====================================================================
template <int AMODE, int BT, int ACT>
__global__ __launch_bounds__(256) void gemm_wmma_k(
    const float* __restrict__ A, const float* __restrict__ Bm,
    const float* __restrict__ bias, float* __restrict__ Cm,
    int M, int N, int K, int ldc,
    const float* __restrict__ im_src, int im_cin, int im_in, int im_out)
{
  __shared__ _Float16 As[2][64][40];   // 2 x (64x32 tile, padded)
  __shared__ _Float16 Bs[2][32][72];   // 2 x (32x64 tile, padded)

  const int tid  = threadIdx.x;
  const int wid  = tid >> 5, lane = tid & 31;
  const int wm   = wid >> 1, wn   = wid & 1;
  const int m0   = blockIdx.y * 64, n0 = blockIdx.x * 64;

  const int o2 = im_out * im_out, o3 = o2 * im_out;
  const int i2 = im_in * im_in;

  // staging coordinates: 2 float4 per thread for each of A and B
  const int ar  = tid >> 3;         // A rows ar and ar+32
  const int ac  = (tid & 7) * 4;    // A col group
  const int brr = tid >> 4;         // B rows brr and brr+16 (BT==0)
  const int bcc = (tid & 15) * 4;   // B col group           (BT==0)
  const int tn  = tid & 63;         // BT==1: n index; k rows (tid>>6)*4 (+16)

  v8f acc0 = {0.f, 0.f, 0.f, 0.f, 0.f, 0.f, 0.f, 0.f};
  v8f acc1 = {0.f, 0.f, 0.f, 0.f, 0.f, 0.f, 0.f, 0.f};

  auto stageA = [&](int k0, float4& a0, float4& a1) {
    if (AMODE == 0) {
      if (m0 + 64 <= M && k0 + 32 <= K) {
        const float* base = A + (size_t)(m0 + ar) * K + k0 + ac;
        a0 = *reinterpret_cast<const float4*>(base);
        a1 = *reinterpret_cast<const float4*>(base + (size_t)32 * K);
      } else {
        float t0[4], t1[4];
#pragma unroll
        for (int i = 0; i < 4; ++i) {
          int k = k0 + ac + i;
          t0[i] = (m0 + ar      < M && k < K) ? A[(size_t)(m0 + ar)      * K + k] : 0.f;
          t1[i] = (m0 + ar + 32 < M && k < K) ? A[(size_t)(m0 + ar + 32) * K + k] : 0.f;
        }
        a0 = make_float4(t0[0], t0[1], t0[2], t0[3]);
        a1 = make_float4(t1[0], t1[1], t1[2], t1[3]);
      }
    } else {
      float t[8];
#pragma unroll
      for (int q = 0; q < 2; ++q)
#pragma unroll
        for (int i = 0; i < 4; ++i) {
          int m = m0 + ar + q * 32, k = k0 + ac + i;
          float v = 0.f;
          if (m < M && k < K) {
            int b  = m / o3;      int s  = m - b * o3;
            int oz = s / o2;      int t2 = s - oz * o2;
            int oy = t2 / im_out; int ox = t2 - oy * im_out;
            int ci = k / 27;      int u  = k - ci * 27;
            int kz = u / 9;       int v9 = u - kz * 9;
            int ky = v9 / 3;      int kx = v9 - ky * 3;
            v = im_src[(size_t)((b * im_cin + ci) * im_in + oz + kz) * i2 +
                       (oy + ky) * im_in + (ox + kx)];
          }
          t[q * 4 + i] = v;
        }
      a0 = make_float4(t[0], t[1], t[2], t[3]);
      a1 = make_float4(t[4], t[5], t[6], t[7]);
    }
  };

  auto stageB = [&](int k0, float4& b0, float4& b1) {
    if (BT == 0) {
      if (k0 + 32 <= K && n0 + 64 <= N) {
        const float* base = Bm + (size_t)(k0 + brr) * N + n0 + bcc;
        b0 = *reinterpret_cast<const float4*>(base);
        b1 = *reinterpret_cast<const float4*>(base + (size_t)16 * N);
      } else {
        float t0[4], t1[4];
#pragma unroll
        for (int i = 0; i < 4; ++i) {
          int n = n0 + bcc + i;
          t0[i] = (k0 + brr      < K && n < N) ? Bm[(size_t)(k0 + brr)      * N + n] : 0.f;
          t1[i] = (k0 + brr + 16 < K && n < N) ? Bm[(size_t)(k0 + brr + 16) * N + n] : 0.f;
        }
        b0 = make_float4(t0[0], t0[1], t0[2], t0[3]);
        b1 = make_float4(t1[0], t1[1], t1[2], t1[3]);
      }
    } else {  // B given as [N,K]; contiguous along k
      float t[8];
      int n = n0 + tn;
#pragma unroll
      for (int q = 0; q < 2; ++q) {
        int r4 = (tid >> 6) * 4 + q * 16;
#pragma unroll
        for (int i = 0; i < 4; ++i) {
          int k = k0 + r4 + i;
          t[q * 4 + i] = (n < N && k < K) ? Bm[(size_t)n * K + k] : 0.f;
        }
      }
      b0 = make_float4(t[0], t[1], t[2], t[3]);
      b1 = make_float4(t[4], t[5], t[6], t[7]);
    }
  };

  auto storeLDS = [&](int buf, float4 a0, float4 a1, float4 b0, float4 b1) {
    st4h(&As[buf][ar][ac],      a0);
    st4h(&As[buf][ar + 32][ac], a1);
    if (BT == 0) {
      st4h(&Bs[buf][brr][bcc],      b0);
      st4h(&Bs[buf][brr + 16][bcc], b1);
    } else {
      int r4a = (tid >> 6) * 4, r4b = r4a + 16;
      Bs[buf][r4a + 0][tn] = (_Float16)b0.x;
      Bs[buf][r4a + 1][tn] = (_Float16)b0.y;
      Bs[buf][r4a + 2][tn] = (_Float16)b0.z;
      Bs[buf][r4a + 3][tn] = (_Float16)b0.w;
      Bs[buf][r4b + 0][tn] = (_Float16)b1.x;
      Bs[buf][r4b + 1][tn] = (_Float16)b1.y;
      Bs[buf][r4b + 2][tn] = (_Float16)b1.z;
      Bs[buf][r4b + 3][tn] = (_Float16)b1.w;
    }
  };

  auto compute = [&](int buf) {
    // A frag (16x32): row per lane, K split by lane-half (ISA 7.12.2)
    const int r  = lane & 15, kh = lane >> 4;
    v16h a;
#pragma unroll
    for (int i = 0; i < 8; ++i) a[i]     = As[buf][wm * 16 + r][kh * 8 + i];
#pragma unroll
    for (int i = 0; i < 8; ++i) a[8 + i] = As[buf][wm * 16 + r][16 + kh * 8 + i];
    // B frags (32x16): lane holds column n = lane&15; k in same split as A
    const int kb = kh << 3;
    v16h b0v, b1v;
#pragma unroll
    for (int i = 0; i < 16; ++i) b0v[i] = Bs[buf][kb + (i & 7) + ((i >> 3) << 4)][wn * 32 + r];
#pragma unroll
    for (int i = 0; i < 16; ++i) b1v[i] = Bs[buf][kb + (i & 7) + ((i >> 3) << 4)][wn * 32 + 16 + r];
    acc0 = __builtin_amdgcn_wmma_f32_16x16x32_f16(false, a, false, b0v, (short)0, acc0, false, false);
    acc1 = __builtin_amdgcn_wmma_f32_16x16x32_f16(false, a, false, b1v, (short)0, acc1, false, false);
  };

  // ---- software-pipelined main loop (double-buffered LDS) ----
  const int nk = (K + 31) >> 5;
  {
    float4 a0, a1, b0, b1;
    stageA(0, a0, a1);
    stageB(0, b0, b1);
    storeLDS(0, a0, a1, b0, b1);
  }
  __syncthreads();
  for (int kt = 0; kt < nk; ++kt) {
    float4 na0, na1, nb0, nb1;
    const bool more = (kt + 1 < nk);
    if (more) {
      stageA((kt + 1) << 5, na0, na1);
      stageB((kt + 1) << 5, nb0, nb1);
      if (AMODE == 0 && kt + 2 < nk && m0 + ar < M)
        __builtin_prefetch(A + (size_t)(m0 + ar) * K + ((kt + 2) << 5) + ac, 0, 1);
    }
    compute(kt & 1);
    if (more) storeLDS((kt + 1) & 1, na0, na1, nb0, nb1);
    __syncthreads();
  }

  // ---- writeback: C/D layout (VGPR i -> M = i + 8*(lane>=16), N = lane&15) ----
  const int nlo = lane & 15, mhi = (lane >> 4) * 8;
#pragma unroll
  for (int i = 0; i < 8; ++i) {
    int row = m0 + wm * 16 + mhi + i;
    if (row >= M) continue;
    int c0 = n0 + wn * 32 + nlo;
    if (c0 < N) {
      float v = acc0[i] + (bias ? bias[c0] : 0.f);
      if (ACT) v = leakyf(v, 0.01f);
      Cm[(size_t)row * ldc + c0] = v;
    }
    int c1 = c0 + 16;
    if (c1 < N) {
      float v = acc1[i] + (bias ? bias[c1] : 0.f);
      if (ACT) v = leakyf(v, 0.01f);
      Cm[(size_t)row * ldc + c1] = v;
    }
  }
}

// =====================================================================
// conv1: direct conv(2->32, 3^3) + bias + BN(eval) + leaky + 2x2x2 maxpool
// in  [32][2][50][50][50], out [32][32][24][24][24]
// =====================================================================
__global__ void conv1_fused_k(const float* __restrict__ x, const float* __restrict__ w,
                              const float* __restrict__ cb, const float* __restrict__ g,
                              const float* __restrict__ bt, float* __restrict__ out, int total)
{
  int tid = blockIdx.x * blockDim.x + threadIdx.x;
  if (tid >= total) return;
  int px = tid % 24; int t = tid / 24;
  int py = t % 24; t /= 24;
  int pz = t % 24; t /= 24;
  int co = t % 32; int b = t / 32;
  float scale = g[co] * rsqrtf(1.f + 1e-5f);
  float beta  = bt[co];
  float bias  = cb[co];
  const float* wc = w + co * 54;
  float mx = -3.4e38f;
  for (int dz = 0; dz < 2; ++dz)
    for (int dy = 0; dy < 2; ++dy)
      for (int dx = 0; dx < 2; ++dx) {
        int oz = 2 * pz + dz, oy = 2 * py + dy, ox = 2 * px + dx;
        float acc = bias;
        for (int ci = 0; ci < 2; ++ci) {
          const float* xin = x + (size_t)(b * 2 + ci) * 125000;
          const float* wk  = wc + ci * 27;
#pragma unroll
          for (int kz = 0; kz < 3; ++kz)
#pragma unroll
            for (int ky = 0; ky < 3; ++ky)
#pragma unroll
              for (int kx = 0; kx < 3; ++kx)
                acc += xin[(size_t)(oz + kz) * 2500 + (oy + ky) * 50 + (ox + kx)] *
                       wk[kz * 9 + ky * 3 + kx];
        }
        mx = fmaxf(mx, leakyf(acc * scale + beta, 0.01f));
      }
  out[tid] = mx;
}

// conv2 epilogue: pre [(b,oz,oy,ox)][64] (spatial 22) -> BN + leaky + pool ->
// flat [32][64*11*11*11] in C,D,H,W order
__global__ void pool2_k(const float* __restrict__ pre, const float* __restrict__ g,
                        const float* __restrict__ bt, float* __restrict__ out, int total)
{
  int tid = blockIdx.x * blockDim.x + threadIdx.x;
  if (tid >= total) return;
  int x = tid % 11; int t = tid / 11;
  int y = t % 11; t /= 11;
  int z = t % 11; t /= 11;
  int co = t % 64; int b = t / 64;
  float scale = g[co] * rsqrtf(1.f + 1e-5f);
  float beta  = bt[co];
  float mx = -3.4e38f;
  for (int dz = 0; dz < 2; ++dz)
    for (int dy = 0; dy < 2; ++dy)
      for (int dx = 0; dx < 2; ++dx) {
        int m = b * 10648 + (2 * z + dz) * 484 + (2 * y + dy) * 22 + (2 * x + dx);
        mx = fmaxf(mx, leakyf(pre[(size_t)m * 64 + co] * scale + beta, 0.01f));
      }
  out[tid] = mx;
}

// ---------------------- small utility kernels ----------------------
__global__ void fill_k(float* p, float v, int n)
{
  int i = blockIdx.x * blockDim.x + threadIdx.x;
  if (i < n) p[i] = v;
}

__global__ void init_bias_k(float* out, const float* b1, const float* b2, int rows, int cols)
{
  int i = blockIdx.x * blockDim.x + threadIdx.x;
  if (i >= rows * cols) return;
  int c = i % cols;
  out[i] = b1[c] + (b2 ? b2[c] : 0.f);
}

__global__ void build_aug_k(const int* s, const int* d, int E, int Nd, int* so, int* dout)
{
  int i = blockIdx.x * blockDim.x + threadIdx.x;
  if (i < E) { so[i] = s[i]; dout[i] = d[i]; }
  else if (i < E + Nd) { so[i] = i - E; dout[i] = i - E; }
}

__global__ void cnt_k(const int* d, int E, float* cnt)
{
  int e = blockIdx.x * blockDim.x + threadIdx.x;
  if (e < E) atomicAdd(&cnt[d[e]], 1.f);
}

__global__ void sum_k(const float* __restrict__ ea, const int* __restrict__ d, int E,
                      float* __restrict__ sum)
{
  int i = blockIdx.x * blockDim.x + threadIdx.x;
  if (i >= E * 64) return;
  int e = i >> 6, q = (i & 63) * 4;
  float4 v = *reinterpret_cast<const float4*>(ea + (size_t)e * 256 + q);
  float* sp = sum + (size_t)d[e] * 256 + q;
  atomicAdd(sp + 0, v.x); atomicAdd(sp + 1, v.y);
  atomicAdd(sp + 2, v.z); atomicAdd(sp + 3, v.w);
}

__global__ void loopwrite_k(const float* sum, const float* cnt, float* eaA, int E, int Nd)
{
  int i = blockIdx.x * blockDim.x + threadIdx.x;
  if (i >= Nd * 256) return;
  int node = i >> 8;
  eaA[(size_t)(E + node) * 256 + (i & 255)] = sum[i] / fmaxf(cnt[node], 1.f);
}

// ---------------------- GATv2 attention kernels ----------------------
// one wave per (edge, head); shuffle-reduced dot with att vector
__global__ __launch_bounds__(256) void gat_score_k(
    const float* __restrict__ hs, const float* __restrict__ hd, const float* __restrict__ he,
    const int* __restrict__ srcl, const int* __restrict__ dstl,
    const float* __restrict__ att, float* __restrict__ score, int e0, int ec, int C)
{
  int wid = threadIdx.x >> 5, lane = threadIdx.x & 31;
  int g = blockIdx.x * 8 + wid;
  if (g >= ec * 2) return;
  int e = e0 + (g >> 1), h = g & 1;
  int C2 = 2 * C;
  const float* hsr = hs + (size_t)srcl[e] * C2 + h * C;
  const float* hdr = hd + (size_t)dstl[e] * C2 + h * C;
  const float* her = he + (size_t)(e - e0) * C2 + h * C;
  const float* at  = att + h * C;
  float s = 0.f;
  for (int c = lane; c < C; c += 32)
    s += at[c] * leakyf(hsr[c] + hdr[c] + her[c], 0.2f);
#pragma unroll
  for (int off = 16; off > 0; off >>= 1) s += __shfl_xor(s, off, 32);
  if (lane == 0) score[(size_t)e * 2 + h] = s;
}

__device__ void atomicMaxF(float* a, float v)
{
  unsigned int* u = (unsigned int*)a;
  unsigned int old = __float_as_uint(*a);
  while (__uint_as_float(old) < v) {
    unsigned int assumed = old;
    old = atomicCAS(u, assumed, __float_as_uint(v));
    if (old == assumed) break;
  }
}

__global__ void segmax_k(const float* score, const int* dstl, float* segmax, int Ea)
{
  int i = blockIdx.x * blockDim.x + threadIdx.x;
  if (i >= Ea * 2) return;
  atomicMaxF(&segmax[dstl[i >> 1] * 2 + (i & 1)], score[i]);
}

__global__ void fixmax_k(float* segmax, int n)
{
  int i = blockIdx.x * blockDim.x + threadIdx.x;
  if (i < n) { float m = segmax[i]; if (!(m > -1e29f)) segmax[i] = 0.f; }
}

__global__ void expsum_k(float* score, const int* dstl, const float* segmax, float* segsum, int Ea)
{
  int i = blockIdx.x * blockDim.x + threadIdx.x;
  if (i >= Ea * 2) return;
  int idx = dstl[i >> 1] * 2 + (i & 1);
  float ev = __expf(score[i] - segmax[idx]);
  score[i] = ev;
  atomicAdd(&segsum[idx], ev);
}

__global__ void norm_k(float* score, const int* dstl, const float* segsum, int Ea)
{
  int i = blockIdx.x * blockDim.x + threadIdx.x;
  if (i >= Ea * 2) return;
  score[i] = score[i] / (segsum[dstl[i >> 1] * 2 + (i & 1)] + 1e-16f);
}

__global__ void agg_k(const float* __restrict__ hs, const float* __restrict__ alpha,
                      const int* __restrict__ srcl, const int* __restrict__ dstl,
                      float* __restrict__ out, int Ea, int C)
{
  int i = blockIdx.x * blockDim.x + threadIdx.x;
  int C2 = 2 * C, qn = C2 >> 2;
  if (i >= Ea * qn) return;
  int e = i / qn, q = (i - e * qn) * 4;
  float a = alpha[(size_t)e * 2 + (q >= C ? 1 : 0)];
  float4 v = *reinterpret_cast<const float4*>(hs + (size_t)srcl[e] * C2 + q);
  float* op = out + (size_t)dstl[e] * C2 + q;
  atomicAdd(op + 0, a * v.x); atomicAdd(op + 1, a * v.y);
  atomicAdd(op + 2, a * v.z); atomicAdd(op + 3, a * v.w);
}

// final: out[32] = xp2[32,1024] @ w[1024,1] + b
__global__ void final_k(const float* __restrict__ xp, const float* __restrict__ w,
                        const float* __restrict__ b, float* __restrict__ out)
{
  __shared__ float red[256];
  int r = blockIdx.x;
  float s = 0.f;
  for (int k = threadIdx.x; k < 1024; k += 256) s += xp[r * 1024 + k] * w[k];
  red[threadIdx.x] = s;
  __syncthreads();
  for (int st = 128; st > 0; st >>= 1) {
    if (threadIdx.x < st) red[threadIdx.x] += red[threadIdx.x + st];
    __syncthreads();
  }
  if (threadIdx.x == 0) out[r] = red[0] + b[0];
}

// =====================================================================
// Host orchestration
// =====================================================================
static inline int cdiv_i(int a, int b) { return (a + b - 1) / b; }

extern "C" void kernel_launch(void* const* d_in, const int* in_sizes, int n_in,
                              void* d_out, int out_size, void* d_ws, size_t ws_size,
                              hipStream_t stream)
{
  (void)in_sizes; (void)n_in; (void)out_size; (void)ws_size;

  auto p = [&](int i) -> const float* { return (const float*)d_in[i]; };

  // inputs
  const float* x_branch = p(0);       // [10000,32]
  const float* x_prop   = p(1);       // [32,34]
  const float* x_patch  = p(2);       // [32,2,50,50,50]
  const int* ei_pp_s = (const int*)d_in[3]; const int* ei_pp_d = ei_pp_s + 256;
  const int* ei_bp_s = (const int*)d_in[4]; const int* ei_bp_d = ei_bp_s + 1024;
  const int* ei_bb_s = (const int*)d_in[5]; const int* ei_bb_d = ei_bb_s + 65536;
  const float* ea_pp = p(6);          // [256,16]
  const float* ea_bp = p(7);          // [1024,16]
  const float* ea_bb = p(8);          // [65536,16]

  // params (setup_inputs insertion order, recursively flattened)
  // node_branch 9..12, node_proposal 13..16, edge_pp 17..20, edge_bp 21..24, edge_bb 25..28
  // conv: c1w 29, c1b 30, bn1g 31, bn1b 32, c2w 33, c2b 34, bn2g 35, bn2b 36,
  //       f1w 37, f1b 38, f2w 39, f2b 40
  struct GatP { const float *wl, *bl, *wr, *br, *we, *att, *bias; };
  auto GP = [&](int base) {
    return GatP{p(base), p(base + 1), p(base + 2), p(base + 3), p(base + 4), p(base + 5), p(base + 6)};
  };
  GatP g1pp = GP(41), g1bp = GP(48), g1bb = GP(55);
  GatP g2pp = GP(62), g2bp = GP(69), g2bb = GP(76);
  const float* out_w = p(83);
  const float* out_b = p(84);

  // ---------------- workspace bump allocator ----------------
  char* ws = (char*)d_ws;
  size_t off = 0;
  auto falloc = [&](size_t elems) -> float* {
    float* q = (float*)(ws + off);
    off += ((elems * 4 + 255) / 256) * 256;
    return q;
  };
  auto ialloc = [&](size_t elems) -> int* {
    int* q = (int*)(ws + off);
    off += ((elems * 4 + 255) / 256) * 256;
    return q;
  };

  float* conv1p   = falloc(14155776);              // [32,32,24,24,24]
  float* conv2pre = falloc((size_t)340736 * 64);
  float* flat     = falloc((size_t)32 * 85184);
  float* fc1      = falloc(32 * 256);
  float* hbig     = falloc((size_t)65536 * 512);   // shared MLP hidden scratch
  float* xp0      = falloc(32 * 256);
  float* xb0      = falloc((size_t)10000 * 256);
  float* eaApp    = falloc((size_t)288 * 256);
  float* eaAbp    = falloc((size_t)1024 * 256);
  float* eaAbb    = falloc((size_t)75536 * 256);
  int*   spp      = ialloc(288);
  int*   dpp      = ialloc(288);
  int*   sbb      = ialloc(75536);
  int*   dbb      = ialloc(75536);
  float* cntb     = falloc(10000);
  float* sumb     = falloc((size_t)10000 * 256);
  float* hsbuf    = falloc((size_t)10000 * 1024);
  float* hdbuf    = falloc((size_t)10000 * 1024);
  float* hebuf    = falloc((size_t)8192 * 1024);
  float* score    = falloc((size_t)75536 * 2);
  float* segmax   = falloc(10000 * 2);
  float* segsum   = falloc(10000 * 2);
  float* xp1      = falloc(32 * 512);
  float* xb1      = falloc((size_t)10000 * 512);
  float* xp2      = falloc(32 * 1024);
  float* xb2      = falloc((size_t)10000 * 1024);

  auto fill = [&](float* q, int n, float v) {
    fill_k<<<cdiv_i(n, 256), 256, 0, stream>>>(q, v, n);
  };
  auto gemm = [&](const float* A, const float* Bm, const float* bias, float* Cm,
                  int M, int N, int K, int ldc, int act) {
    dim3 grid((unsigned)cdiv_i(N, 64), (unsigned)cdiv_i(M, 64));
    if (act)
      gemm_wmma_k<0, 0, 1><<<grid, 256, 0, stream>>>(A, Bm, bias, Cm, M, N, K, ldc, nullptr, 1, 1, 1);
    else
      gemm_wmma_k<0, 0, 0><<<grid, 256, 0, stream>>>(A, Bm, bias, Cm, M, N, K, ldc, nullptr, 1, 1, 1);
  };

  // ---------------- ConvNet ----------------
  conv1_fused_k<<<cdiv_i(14155776, 256), 256, 0, stream>>>(
      x_patch, p(29), p(30), p(31), p(32), conv1p, 14155776);
  {
    // conv2 as implicit GEMM: M = 32*22^3 = 340736, N = 64, K = 864
    dim3 grid(1, (unsigned)cdiv_i(340736, 64));
    gemm_wmma_k<1, 1, 0><<<grid, 256, 0, stream>>>(
        nullptr, p(33), p(34), conv2pre, 340736, 64, 864, 64, conv1p, 32, 24, 22);
  }
  pool2_k<<<cdiv_i(2725888, 256), 256, 0, stream>>>(conv2pre, p(35), p(36), flat, 2725888);
  gemm(flat, p(37), p(38), fc1, 32, 256, 85184, 256, 1);     // fc1 + leaky
  gemm(fc1, p(39), p(40), xp0 + 128, 32, 128, 256, 256, 0);  // -> xp0[:,128:256]

  // ---------------- node / edge MLPs ----------------
  gemm(x_prop, p(13), p(14), hbig, 32, 256, 34, 256, 1);
  gemm(hbig, p(15), p(16), xp0, 32, 128, 256, 256, 0);       // -> xp0[:,0:128]

  gemm(x_branch, p(9), p(10), hbig, 10000, 512, 32, 512, 1);
  gemm(hbig, p(11), p(12), xb0, 10000, 256, 512, 256, 0);

  gemm(ea_pp, p(17), p(18), hbig, 256, 512, 16, 512, 1);
  gemm(hbig, p(19), p(20), eaApp, 256, 256, 512, 256, 0);
  gemm(ea_bp, p(21), p(22), hbig, 1024, 512, 16, 512, 1);
  gemm(hbig, p(23), p(24), eaAbp, 1024, 256, 512, 256, 0);
  gemm(ea_bb, p(25), p(26), hbig, 65536, 512, 16, 512, 1);
  gemm(hbig, p(27), p(28), eaAbb, 65536, 256, 512, 256, 0);

  // ---------------- self-loop edge-attr means + augmented edge lists ----------------
  // pp: E=256, Nd=32
  fill(cntb, 32, 0.f); fill(sumb, 32 * 256, 0.f);
  cnt_k<<<1, 256, 0, stream>>>(ei_pp_d, 256, cntb);
  sum_k<<<cdiv_i(256 * 64, 256), 256, 0, stream>>>(eaApp, ei_pp_d, 256, sumb);
  loopwrite_k<<<cdiv_i(32 * 256, 256), 256, 0, stream>>>(sumb, cntb, eaApp, 256, 32);
  build_aug_k<<<cdiv_i(288, 256), 256, 0, stream>>>(ei_pp_s, ei_pp_d, 256, 32, spp, dpp);
  // bb: E=65536, Nd=10000
  fill(cntb, 10000, 0.f); fill(sumb, 10000 * 256, 0.f);
  cnt_k<<<cdiv_i(65536, 256), 256, 0, stream>>>(ei_bb_d, 65536, cntb);
  sum_k<<<cdiv_i(65536 * 64, 256), 256, 0, stream>>>(eaAbb, ei_bb_d, 65536, sumb);
  loopwrite_k<<<cdiv_i(10000 * 256, 256), 256, 0, stream>>>(sumb, cntb, eaAbb, 65536, 10000);
  build_aug_k<<<cdiv_i(75536, 256), 256, 0, stream>>>(ei_bb_s, ei_bb_d, 65536, 10000, sbb, dbb);

  // ---------------- GATv2 layer runner ----------------
  const int CH = 8192;
  auto run_gat = [&](const float* xs, int Ns, int Ds, const float* xd, int Nd_, int Dd,
                     const int* srcl, const int* dstl, int Ea, const float* eaA,
                     const GatP& gp, int C, float* outBuf) {
    int C2 = 2 * C;
    gemm(xs, gp.wl, gp.bl, hsbuf, Ns, C2, Ds, C2, 0);
    gemm(xd, gp.wr, gp.br, hdbuf, Nd_, C2, Dd, C2, 0);
    fill(segmax, Nd_ * 2, -1e30f);
    fill(segsum, Nd_ * 2, 0.f);
    for (int e0 = 0; e0 < Ea; e0 += CH) {
      int ec = (Ea - e0 < CH) ? (Ea - e0) : CH;
      gemm(eaA + (size_t)e0 * 256, gp.we, nullptr, hebuf, ec, C2, 256, C2, 0);
      gat_score_k<<<cdiv_i(ec * 2, 8), 256, 0, stream>>>(
          hsbuf, hdbuf, hebuf, srcl, dstl, gp.att, score, e0, ec, C);
    }
    segmax_k<<<cdiv_i(Ea * 2, 256), 256, 0, stream>>>(score, dstl, segmax, Ea);
    fixmax_k<<<cdiv_i(Nd_ * 2, 256), 256, 0, stream>>>(segmax, Nd_ * 2);
    expsum_k<<<cdiv_i(Ea * 2, 256), 256, 0, stream>>>(score, dstl, segmax, segsum, Ea);
    norm_k<<<cdiv_i(Ea * 2, 256), 256, 0, stream>>>(score, dstl, segsum, Ea);
    int qn = C2 / 4;
    agg_k<<<cdiv_i(Ea * qn, 256), 256, 0, stream>>>(hsbuf, score, srcl, dstl, outBuf, Ea, C);
  };

  // ---------------- layer 1 (C = 256) ----------------
  init_bias_k<<<cdiv_i(32 * 512, 256), 256, 0, stream>>>(xp1, g1pp.bias, g1bp.bias, 32, 512);
  init_bias_k<<<cdiv_i(10000 * 512, 256), 256, 0, stream>>>(xb1, g1bb.bias, nullptr, 10000, 512);
  run_gat(xp0, 32, 256, xp0, 32, 256, spp, dpp, 288, eaApp, g1pp, 256, xp1);
  run_gat(xb0, 10000, 256, xp0, 32, 256, ei_bp_s, ei_bp_d, 1024, eaAbp, g1bp, 256, xp1);
  run_gat(xb0, 10000, 256, xb0, 10000, 256, sbb, dbb, 75536, eaAbb, g1bb, 256, xb1);

  // ---------------- layer 2 (C = 512) ----------------
  init_bias_k<<<cdiv_i(32 * 1024, 256), 256, 0, stream>>>(xp2, g2pp.bias, g2bp.bias, 32, 1024);
  init_bias_k<<<cdiv_i(10000 * 1024, 256), 256, 0, stream>>>(xb2, g2bb.bias, nullptr, 10000, 1024);
  run_gat(xp1, 32, 512, xp1, 32, 512, spp, dpp, 288, eaApp, g2pp, 512, xp2);
  run_gat(xb1, 10000, 512, xp1, 32, 512, ei_bp_s, ei_bp_d, 1024, eaAbp, g2bp, 512, xp2);
  run_gat(xb1, 10000, 512, xb1, 10000, 512, sbb, dbb, 75536, eaAbb, g2bb, 512, xb2);

  // ---------------- output head ----------------
  final_k<<<32, 256, 0, stream>>>(xp2, out_w, out_b, (float*)d_out);
}
